// CrossCoder_76381698392290
// MI455X (gfx1250) — matrix operand order repaired and checked by reading, not compiled, not run on hardware
//
#include <hip/hip_runtime.h>
#include <stdint.h>

// ---------------- problem constants ----------------
#define B_SZ 2048
#define D_SZ 2048
#define F_SZ 32768
#define K_TOP 64
#define KD 4096            // M*D contraction dim

// ---------------- WMMA types (CDNA5 / gfx1250) ----------------
typedef __attribute__((ext_vector_type(16))) __bf16 v16bf;
typedef __attribute__((ext_vector_type(8)))  __bf16 v8bf;
typedef __attribute__((ext_vector_type(2)))  __bf16 v2bf;
typedef __attribute__((ext_vector_type(8)))  float  v8f;

#if defined(__has_builtin)
#if __has_builtin(__builtin_amdgcn_cvt_pk_bf16_f32)
#define HAVE_CVT_PK_BF16 1
#endif
#endif

__device__ __forceinline__ unsigned short f2bf(float f) {
    uint32_t u = __float_as_uint(f);
    // round-to-nearest-even fp32 -> bf16
    return (unsigned short)((u + 0x7FFFu + ((u >> 16) & 1u)) >> 16);
}
__device__ __forceinline__ uint32_t f2bf_pk(float lo, float hi) {
#ifdef HAVE_CVT_PK_BF16
    union { v2bf v; uint32_t u; } c;
    c.v = __builtin_amdgcn_cvt_pk_bf16_f32(lo, hi);   // v_cvt_pk_bf16_f32 (RNE)
    return c.u;
#else
    return (uint32_t)f2bf(lo) | ((uint32_t)f2bf(hi) << 16);
#endif
}

// ================= encode GEMM: pre = relu(X @ W_enc + b_enc) =================
// X: (2048 x 4096) row-major fp32; W_enc: (4096 x 32768) row-major fp32
// block tile 128x128, K-step 32, 256 threads = 8 waves (wave32), waves in 4x2,
// each wave does a 32x64 sub-tile = 2x4 fragments of v_wmma_f32_16x16x32_bf16.
// Double-buffered LDS: one barrier per K-step, global loads overlapped with WMMA.
#define MT 128
#define NT 128
#define KT 32
#define LDSTR 40            // padded LDS row stride in ushorts (80B = 20 banks)
#define ABUF (MT * LDSTR)   // one A buffer, ushorts
#define BBUF (NT * LDSTR)   // one B buffer, ushorts

struct TileRegs {
    float4 x[4];   // X rows (xr + 32*rr), 4 consecutive k
    float4 w[4];   // W rows: [pass*2 + r] = row (pass*16 + wk2*2 + r), 4 consecutive n
};

__device__ __forceinline__ void load_tile(
    const float* __restrict__ X, const float* __restrict__ W,
    int rowBase, int colBase, int k0, int xr, int xq, int wq, int wk2,
    TileRegs& t)
{
#pragma unroll
    for (int rr = 0; rr < 4; rr++)
        t.x[rr] = *(const float4*)(X + (size_t)(rowBase + xr + rr * 32) * KD + k0 + xq * 4);
#pragma unroll
    for (int pass = 0; pass < 2; pass++)
#pragma unroll
        for (int r = 0; r < 2; r++)
            t.w[pass * 2 + r] = *(const float4*)(
                W + (size_t)(k0 + pass * 16 + wk2 * 2 + r) * F_SZ + colBase + wq * 4);
}

__device__ __forceinline__ void store_tile(
    unsigned short* lA, unsigned short* lB,
    int xr, int xq, int wq, int wk2, const TileRegs& t)
{
    // A: [row][k] bf16, two packed dwords per row chunk (8B ds_store)
#pragma unroll
    for (int rr = 0; rr < 4; rr++) {
        uint2 s;
        s.x = f2bf_pk(t.x[rr].x, t.x[rr].y);
        s.y = f2bf_pk(t.x[rr].z, t.x[rr].w);
        *(uint2*)(&lA[(xr + rr * 32) * LDSTR + xq * 4]) = s;
    }
    // B: transpose to [col][k] bf16; pack two adjacent k per dword (4B ds_store)
#pragma unroll
    for (int pass = 0; pass < 2; pass++) {
        const float4 v0 = t.w[pass * 2 + 0];
        const float4 v1 = t.w[pass * 2 + 1];
        const int kk = pass * 16 + wk2 * 2;
        *(uint32_t*)(&lB[(wq * 4 + 0) * LDSTR + kk]) = f2bf_pk(v0.x, v1.x);
        *(uint32_t*)(&lB[(wq * 4 + 1) * LDSTR + kk]) = f2bf_pk(v0.y, v1.y);
        *(uint32_t*)(&lB[(wq * 4 + 2) * LDSTR + kk]) = f2bf_pk(v0.z, v1.z);
        *(uint32_t*)(&lB[(wq * 4 + 3) * LDSTR + kk]) = f2bf_pk(v0.w, v1.w);
    }
}

__global__ __launch_bounds__(256) void encode_gemm_kernel(
    const float* __restrict__ X, const float* __restrict__ W,
    const float* __restrict__ b_enc, float* __restrict__ features)
{
    __shared__ __attribute__((aligned(16))) unsigned short lA[2 * ABUF]; // A tiles bf16 [row][k]
    __shared__ __attribute__((aligned(16))) unsigned short lB[2 * BBUF]; // B tiles bf16 [col][k]

    const int tid   = threadIdx.x;
    const int lane  = tid & 31;
    const int wave  = tid >> 5;
    const int rowBase = blockIdx.x * MT;   // over B (x fastest -> W col-strip L2 reuse)
    const int colBase = blockIdx.y * NT;   // over F
    const int mwave = (wave & 3) * 32;
    const int nwave = (wave >> 2) * 64;
    const int half  = lane >> 4;           // lane-half per ISA fragment layout
    const int l16   = lane & 15;

    v8f acc[2][4];
#pragma unroll
    for (int i = 0; i < 2; i++)
#pragma unroll
        for (int j = 0; j < 4; j++)
#pragma unroll
            for (int e = 0; e < 8; e++) acc[i][j][e] = 0.0f;

    // global-load thread mappings
    const int xq  = tid & 7;   // float4 slot within 32-float k-row of X tile
    const int xr  = tid >> 3;  // 0..31 base row
    const int wq  = tid & 31;  // float4 slot along n of W tile
    const int wk2 = tid >> 5;  // 0..7 -> k-pair base

    // prologue: tile 0
    TileRegs tr;
    load_tile(X, W, rowBase, colBase, 0, xr, xq, wq, wk2, tr);
    store_tile(lA, lB, xr, xq, wq, wk2, tr);
    __syncthreads();

    int p = 0;
    for (int k0 = 0; k0 < KD; k0 += KT) {
        const bool more = (k0 + KT) < KD;
        if (more) {
            // issue next tile's global loads early (overlap with WMMA below)
            load_tile(X, W, rowBase, colBase, k0 + KT, xr, xq, wq, wk2, tr);
            if (k0 + 2 * KT < KD) {
                // prefetch the W stream two tiles ahead (global_prefetch_b8)
                __builtin_prefetch(W + (size_t)(k0 + 2 * KT + wk2 * 2) * F_SZ + colBase + wq * 4, 0, 3);
            }
        }

        const unsigned short* Ap = lA + p * ABUF;
        const unsigned short* Bp = lB + p * BBUF;

        // ---- fragment loads per ISA 7.12.2 16-bit layouts ----
        v16bf afrag[2], bfrag[4];
#pragma unroll
        for (int i = 0; i < 2; i++) {
            const int row = mwave + i * 16 + l16;
            const v8bf lo = *(const v8bf*)(&Ap[row * LDSTR + half * 8]);        // K 0..7 / 8..15
            const v8bf hi = *(const v8bf*)(&Ap[row * LDSTR + half * 8 + 16]);   // K 16..23 / 24..31
#pragma unroll
            for (int e = 0; e < 8; e++) { afrag[i][e] = lo[e]; afrag[i][e + 8] = hi[e]; }
        }
#pragma unroll
        for (int j = 0; j < 4; j++) {
            const int col = nwave + j * 16 + l16;
            const v8bf lo = *(const v8bf*)(&Bp[col * LDSTR + half * 8]);        // K 0..7 / 8..15
            const v8bf hi = *(const v8bf*)(&Bp[col * LDSTR + half * 8 + 16]);   // K 16..23 / 24..31
#pragma unroll
            for (int e = 0; e < 8; e++) { bfrag[j][e] = lo[e]; bfrag[j][e + 8] = hi[e]; }
        }

#pragma unroll
        for (int i = 0; i < 2; i++)
#pragma unroll
            for (int j = 0; j < 4; j++)
                acc[i][j] = __builtin_amdgcn_wmma_f32_16x16x32_bf16(
                    false, afrag[i], false, bfrag[j], (short)0, acc[i][j], false, false);

        if (more)
            store_tile(lA + (p ^ 1) * ABUF, lB + (p ^ 1) * BBUF, xr, xq, wq, wk2, tr);
        __syncthreads();
        p ^= 1;
    }

    // ---- epilogue: +bias, relu, store (D layout: VGPR r -> rows r / r+8 per lane half) ----
#pragma unroll
    for (int i = 0; i < 2; i++)
#pragma unroll
        for (int j = 0; j < 4; j++) {
            const int f = colBase + nwave + j * 16 + l16;
            const float bias = b_enc[f];
#pragma unroll
            for (int r = 0; r < 8; r++) {
                const int brow = rowBase + mwave + i * 16 + half * 8 + r;
                float v = acc[i][j][r] + bias;
                v = v > 0.0f ? v : 0.0f;
                features[(size_t)brow * F_SZ + f] = v;
            }
        }
}

// ================= top-K select (K=64 of 32768), row LDS-resident =================
#define TK 512
__global__ __launch_bounds__(TK) void topk_kernel(
    float* __restrict__ features, float* __restrict__ out_vals, int* __restrict__ out_idx)
{
    __shared__ uint32_t rowv[F_SZ];          // 128 KB (of 320 KB/WGP)
    __shared__ uint32_t hist[256];
    __shared__ uint32_t cntGT[TK];
    __shared__ uint32_t cntEQ[TK];
    __shared__ uint32_t sPrefix, sRemaining, sTotalGT;

    const int b = blockIdx.x;
    const int t = threadIdx.x;
    float* frow = features + (size_t)b * F_SZ;

    for (int i = t; i < F_SZ; i += TK) rowv[i] = __float_as_uint(frow[i]);
    __syncthreads();

    // 4-pass radix select on non-negative float bits (unsigned order == float order)
    uint32_t prefix = 0, remaining = K_TOP;
    const int CH = F_SZ / TK;  // 64 elements per thread, strided (bank-conflict-free)
    for (int shift = 24; shift >= 0; shift -= 8) {
        for (int i = t; i < 256; i += TK) hist[i] = 0;
        __syncthreads();
        const uint32_t pmask = (shift == 24) ? 0u : (0xFFFFFFFFu << (shift + 8));
        for (int i = 0; i < CH; i++) {
            const uint32_t u = rowv[i * TK + t];
            if ((u & pmask) == prefix) atomicAdd(&hist[(u >> shift) & 255u], 1u);
        }
        __syncthreads();
        if (t == 0) {
            uint32_t cum = 0; int d = 255;
            for (; d >= 0; d--) {
                if (cum + hist[d] >= remaining) break;
                cum += hist[d];
            }
            if (d < 0) d = 0;
            sPrefix = prefix | ((uint32_t)d << shift);
            sRemaining = remaining - cum;
        }
        __syncthreads();
        prefix = sPrefix; remaining = sRemaining;
    }
    const uint32_t T = prefix;        // K-th largest value's bits
    const uint32_t need_eq = remaining;

    // deterministic scan-based compaction (fixed permutation order, no ordering atomics)
    uint32_t gt = 0, eq = 0;
    for (int i = 0; i < CH; i++) {
        const uint32_t u = rowv[i * TK + t];
        gt += (u > T); eq += (u == T);
    }
    cntGT[t] = gt; cntEQ[t] = eq;
    __syncthreads();
    if (t == 0) {
        uint32_t aG = 0, aE = 0;
        for (int i = 0; i < TK; i++) {
            const uint32_t g = cntGT[i], e = cntEQ[i];
            cntGT[i] = aG; cntEQ[i] = aE;
            aG += g; aE += e;
        }
        sTotalGT = aG;
    }
    __syncthreads();
    const uint32_t baseG = cntGT[t], baseE = cntEQ[t], totalGT = sTotalGT;

    float* vrow = out_vals + (size_t)b * K_TOP;
    int*   irow = out_idx  + (size_t)b * K_TOP;
    uint32_t g2 = 0, e2 = 0;
    for (int i = 0; i < CH; i++) {
        const int idx = i * TK + t;
        const uint32_t u = rowv[idx];
        const float v = __uint_as_float(u);
        bool sel = false;
        if (u > T) {
            const uint32_t slot = baseG + g2; g2++;
            vrow[slot] = v; irow[slot] = idx; sel = true;
        } else if (u == T) {
            const uint32_t r = baseE + e2; e2++;
            if (r < need_eq) {
                const uint32_t slot = totalGT + r;
                vrow[slot] = v; irow[slot] = idx; sel = true;
            }
        }
        frow[idx] = sel ? v : 0.0f;
    }
}

// ================= sparse decode: recon = sum_k val * W_dec[idx] + b_dec =================
__global__ __launch_bounds__(256) void decode_kernel(
    const float* __restrict__ vals, const int* __restrict__ idx,
    const float* __restrict__ W_dec, const float* __restrict__ b_dec,
    float* __restrict__ recon)
{
    __shared__ float sv[K_TOP];
    __shared__ int   si[K_TOP];
    const int b = blockIdx.x;
    const int t = threadIdx.x;
    if (t < K_TOP) { sv[t] = vals[(size_t)b * K_TOP + t]; si[t] = idx[(size_t)b * K_TOP + t]; }
    __syncthreads();

    const int col0 = t * 16;   // 4096 cols / 256 threads
    float a[16];
#pragma unroll
    for (int e = 0; e < 16; e++) a[e] = b_dec[col0 + e];

    for (int k = 0; k < K_TOP; k++) {
        const float v = sv[k];
        const float* w = W_dec + (size_t)si[k] * KD + col0;
#pragma unroll
        for (int q = 0; q < 4; q++) {
            const float4 wv = *(const float4*)(w + q * 4);
            a[q * 4 + 0] += v * wv.x;
            a[q * 4 + 1] += v * wv.y;
            a[q * 4 + 2] += v * wv.z;
            a[q * 4 + 3] += v * wv.w;
        }
    }
    float* out = recon + (size_t)b * KD + col0;
#pragma unroll
    for (int q = 0; q < 4; q++) {
        float4 o; o.x = a[q*4+0]; o.y = a[q*4+1]; o.z = a[q*4+2]; o.w = a[q*4+3];
        *(float4*)(out + q * 4) = o;
    }
}

// ================= launcher =================
extern "C" void kernel_launch(void* const* d_in, const int* in_sizes, int n_in,
                              void* d_out, int out_size, void* d_ws, size_t ws_size,
                              hipStream_t stream) {
    const float* x     = (const float*)d_in[0];   // (B, M, D)  == (B, KD)
    const float* W_enc = (const float*)d_in[1];   // (M, D, F)  == (KD, F)
    const float* b_enc = (const float*)d_in[2];   // (F,)
    const float* W_dec = (const float*)d_in[3];   // (F, M, D)  == (F, KD)
    const float* b_dec = (const float*)d_in[4];   // (M, D)     == (KD,)

    float* recon    = (float*)d_out;                       // (B, M, D)
    float* features = recon + (size_t)B_SZ * KD;           // (B, F)
    float* ws_vals  = (float*)d_ws;                        // (B, K)
    int*   ws_idx   = (int*)((char*)d_ws + (size_t)B_SZ * K_TOP * sizeof(float));

    encode_gemm_kernel<<<dim3(B_SZ / MT, F_SZ / NT), 256, 0, stream>>>(x, W_enc, b_enc, features);
    topk_kernel<<<B_SZ, TK, 0, stream>>>(features, ws_vals, ws_idx);
    decode_kernel<<<B_SZ, 256, 0, stream>>>(ws_vals, ws_idx, W_dec, b_dec, recon);
}
